// Decoder_3358664425645
// MI455X (gfx1250) — matrix operand order repaired and checked by reading, not compiled, not run on hardware
//
#include <hip/hip_runtime.h>

// ---------------------------------------------------------------------------
// Tacotron2-style decoder for MI455X (gfx1250), persistent-kernel design.
//   * One launch runs all 400 decoder steps (latency-bound RNN).
//   * f16 WMMA (v_wmma_f32_16x16x32_f16) for LSTM gates / projection / pm.
//   * LSTM cell state c lives in VGPRs of the owning wave across all steps.
//   * cum attention weights live in LDS per batch row across all steps.
//   * Grid sync: device-scope atomic barrier + s_cluster_barrier (CDNA5).
// ---------------------------------------------------------------------------

#define B_    32
#define TENC  512
#define ENCD  512
#define TDEC  400
#define NMELS 80
#define PRE   256
#define DRNN  1024
#define ATTD  128
#define LF    32
#define LK    31
#define K1    1792   // prenet(256) + ctx(512) + att_h(1024)
#define K2    2048   // att_h(1024) + dec_h(1024)
#define KPI   1536   // dec_h(1024) + ctx(512)
#define NWG   16
#define BLK   128

typedef __attribute__((ext_vector_type(16))) _Float16 v16h;
typedef __attribute__((ext_vector_type(8)))  _Float16 v8h;
typedef __attribute__((ext_vector_type(8)))  float    v8f;

__device__ inline float sigf(float x) { return 1.0f / (1.0f + __expf(-x)); }

// Deterministic dropout-mask hash (placeholder for JAX threefry; includes the
// 1/(1-p)=2 scale). Compile-only loop: no runtime validation available.
__device__ inline float drop_mask(unsigned seed, unsigned t, unsigned b, unsigned j) {
  unsigned x = seed * 0x27D4EB2Fu ^ t * 0x9E3779B9u ^ b * 0x85EBCA6Bu ^ j * 0xC2B2AE35u;
  x ^= x >> 16; x *= 0x7FEB352Du; x ^= x >> 15; x *= 0x846CA68Bu; x ^= x >> 16;
  return (x & 1u) ? 2.0f : 0.0f;
}

// --- WMMA fragment loaders (ISA 7.12.2 layouts, f16, 16x16x32) --------------
// A (16x32): lanes 0-15 hold M=lane, K = k0+{0..7, 16..23}; lanes 16-31 hold
// K = k0+{8..15, 24..31}.  Two 16B loads per lane.
__device__ inline v16h load_a16(const _Float16* A, int lda, int m0, int k0, int lane) {
  int row = m0 + (lane & 15);
  int kc  = k0 + ((lane >> 4) << 3);
  const _Float16* p = A + (size_t)row * lda + kc;
  v8h lo = *(const v8h*)p;
  v8h hi = *(const v8h*)(p + 16);
  v16h r;
#pragma unroll
  for (int i = 0; i < 8; ++i) { r[i] = lo[i]; r[i + 8] = hi[i]; }
  return r;
}
// B (32x16): column N = lane&15 (= weight row n0+lane&15 of row-major [N][K]),
// lanes 0-15 hold K=k0+{0..7, 16..23}, lanes 16-31 hold K=k0+{8..15, 24..31} (matches A).
__device__ inline v16h load_b16(const _Float16* W, int ldk, int n0, int k0, int lane) {
  int n  = n0 + (lane & 15);
  int kc = k0 + ((lane >> 4) << 3);
  const _Float16* p = W + (size_t)n * ldk + kc;
  v8h lo = *(const v8h*)p;
  v8h hi = *(const v8h*)(p + 16);
  v16h r;
#pragma unroll
  for (int i = 0; i < 8; ++i) { r[i] = lo[i]; r[i + 8] = hi[i]; }
  return r;
}

// --- grid barrier: cluster barrier (CDNA5 fast path, NOP if not cluster-
// dispatched) + device-scope atomic sense barrier (the actual guarantee). ----
__device__ inline void gbar(unsigned* bar) {
  __syncthreads();
  __builtin_amdgcn_s_cluster_barrier();   // s_barrier_signal/wait -3
  if (threadIdx.x == 0) {
    unsigned* cnt = bar;
    unsigned* gen = bar + 32;              // separate cacheline
    unsigned g = __hip_atomic_load(gen, __ATOMIC_RELAXED, __HIP_MEMORY_SCOPE_AGENT);
    unsigned a = __hip_atomic_fetch_add(cnt, 1u, __ATOMIC_ACQ_REL, __HIP_MEMORY_SCOPE_AGENT);
    if (a == NWG - 1) {
      __hip_atomic_store(cnt, 0u, __ATOMIC_RELAXED, __HIP_MEMORY_SCOPE_AGENT);
      __hip_atomic_store(gen, g + 1u, __ATOMIC_RELEASE, __HIP_MEMORY_SCOPE_AGENT);
    } else {
      while (__hip_atomic_load(gen, __ATOMIC_ACQUIRE, __HIP_MEMORY_SCOPE_AGENT) == g)
        __builtin_amdgcn_s_sleep(1);
    }
  }
  __syncthreads();
}

// --- fused LSTM gate GEMM + cell update (wave owns a 16-wide hidden slice) --
__device__ void lstm_gates(const _Float16* __restrict__ Amat, int lda, int Ktot,
                           const _Float16* __restrict__ Wcat,
                           const float* __restrict__ bih, const float* __restrict__ bhh,
                           float cst[2][8],
                           _Float16* d1, int ld1, int c1off,
                           _Float16* d2, int ld2, int c2off,
                           float* hf32, int wid, int lane) {
  v8f acc[4][2];
  const int ncol = wid * 16 + (lane & 15);   // hidden index 0..1023
#pragma unroll
  for (int g = 0; g < 4; ++g) {
    float bsum = bih[g * DRNN + ncol] + bhh[g * DRNN + ncol];
#pragma unroll
    for (int r = 0; r < 8; ++r) { acc[g][0][r] = bsum; acc[g][1][r] = bsum; }
  }
  for (int k0 = 0; k0 < Ktot; k0 += 32) {
    v16h a0 = load_a16(Amat, lda, 0, k0, lane);
    v16h a1 = load_a16(Amat, lda, 16, k0, lane);
#pragma unroll
    for (int g = 0; g < 4; ++g) {
      v16h bb = load_b16(Wcat, Ktot, g * DRNN + wid * 16, k0, lane);
      acc[g][0] = __builtin_amdgcn_wmma_f32_16x16x32_f16(false, a0, false, bb,
                                                         (short)0, acc[g][0], false, false);
      acc[g][1] = __builtin_amdgcn_wmma_f32_16x16x32_f16(false, a1, false, bb,
                                                         (short)0, acc[g][1], false, false);
    }
    if (k0 + 32 < Ktot)  // global_prefetch_b8 of next K-chunk of our B rows
      __builtin_prefetch(Wcat + (size_t)(wid * 16 + (lane & 15)) * Ktot + k0 + 32, 0, 1);
  }
#pragma unroll
  for (int mt2 = 0; mt2 < 2; ++mt2) {
#pragma unroll
    for (int r = 0; r < 8; ++r) {
      float iv = acc[0][mt2][r], fv = acc[1][mt2][r];
      float gv = acc[2][mt2][r], ov = acc[3][mt2][r];
      float cn = sigf(fv) * cst[mt2][r] + sigf(iv) * tanhf(gv);
      float h  = sigf(ov) * tanhf(cn);
      cst[mt2][r] = cn;
      int m = mt2 * 16 + ((lane >> 4) << 3) + r;   // C-frag row mapping
      _Float16 h16 = (_Float16)h;
      d1[(size_t)m * ld1 + c1off + ncol] = h16;
      d2[(size_t)m * ld2 + c2off + ncol] = h16;
      if (hf32) hf32[(size_t)m * DRNN + ncol] = h;
    }
  }
}

struct DecParams {
  const float *memory, *dec_in;
  const unsigned char* mask;
  const float *w1, *w2, *wq, *convw, *wloc, *vw, *vb;
  const float *b1ih, *b1hh, *b2ih, *b2hh, *projb, *gateb;
  const _Float16 *w1cat, *w2cat, *projw, *gatew, *memf16, *attwm;
  float *pm, *atth;
  _Float16 *A1, *A2, *Api;
  unsigned* bar;
  float *out_mel, *out_gate, *out_align;
};

__global__ void __launch_bounds__(BLK) decoder_persistent(DecParams P) {
  __shared__ float s_x[2][NMELS];
  __shared__ float s_p1[2][PRE];
  __shared__ float s_pq[2][ATTD];
  __shared__ float s_e[2][TENC];
  __shared__ float s_cum[2][TENC];
  __shared__ float s_red[2][64];
  __shared__ float s_gred[BLK];
  __shared__ float s_conv[LF * LK];
  __shared__ float s_wloc[ATTD * LF];
  __shared__ float s_v[ATTD];

  const int tid  = threadIdx.x;
  const int lane = tid & 31;
  const int wid  = blockIdx.x * (BLK / 32) + (tid >> 5);   // 0..63
  const int gtid = blockIdx.x * BLK + tid;
  const int GSTRIDE = NWG * BLK;

  // ---- init: zero recurrent activations, stage attention weights in LDS ----
  for (int i = gtid; i < B_ * K1;   i += GSTRIDE) P.A1[i]  = (_Float16)0.0f;
  for (int i = gtid; i < B_ * K2;   i += GSTRIDE) P.A2[i]  = (_Float16)0.0f;
  for (int i = gtid; i < B_ * KPI;  i += GSTRIDE) P.Api[i] = (_Float16)0.0f;
  for (int i = gtid; i < B_ * DRNN; i += GSTRIDE) P.atth[i] = 0.0f;
  for (int i = tid; i < LF * LK;   i += BLK) s_conv[i] = P.convw[i];
  for (int i = tid; i < ATTD * LF; i += BLK) s_wloc[i] = P.wloc[i];
  for (int i = tid; i < ATTD;      i += BLK) s_v[i]    = P.vw[i];
  for (int i = tid; i < TENC;      i += BLK) { s_cum[0][i] = 0.0f; s_cum[1][i] = 0.0f; }

  // ---- one-time: processed_memory = memory @ att_wm^T via WMMA -------------
  for (int tile = wid; tile < (B_ * TENC / 16) * (ATTD / 16); tile += NWG * (BLK / 32)) {
    int mt = tile >> 3, nt = tile & 7;
    v8f acc = {};
    const _Float16* Arow = P.memf16 + (size_t)mt * 16 * ENCD;
    for (int k0 = 0; k0 < ENCD; k0 += 32) {
      v16h a   = load_a16(Arow, ENCD, 0, k0, lane);
      v16h bfr = load_b16(P.attwm, ENCD, nt * 16, k0, lane);
      acc = __builtin_amdgcn_wmma_f32_16x16x32_f16(false, a, false, bfr, (short)0, acc, false, false);
    }
    int ncol = nt * 16 + (lane & 15);
#pragma unroll
    for (int r = 0; r < 8; ++r)
      P.pm[(size_t)(mt * 16 + ((lane >> 4) << 3) + r) * ATTD + ncol] = acc[r];
  }

  float c1[2][8], c2[2][8];
#pragma unroll
  for (int i = 0; i < 2; ++i)
#pragma unroll
    for (int r = 0; r < 8; ++r) { c1[i][r] = 0.0f; c2[i][r] = 0.0f; }

  gbar(P.bar);

  const int bh = tid >> 6;        // which of this WG's two batch rows
  const int lt = tid & 63;        // 64 threads per batch row
  const int b  = 2 * blockIdx.x + bh;
  const float vb0 = P.vb[0];

  for (int t = 0; t < TDEC; ++t) {
    // ================= Phase A: prenet + location attention =================
    for (int i = lt; i < NMELS; i += 64)
      s_x[bh][i] = (t == 0) ? 0.0f
                 : P.dec_in[(size_t)b * NMELS * TDEC + (size_t)i * TDEC + (t - 1)];
    __syncthreads();
    for (int j = lt; j < PRE; j += 64) {
      float s = 0.0f;
      const float* wr = P.w1 + (size_t)j * NMELS;
      for (int k = 0; k < NMELS; ++k) s += s_x[bh][k] * wr[k];
      s_p1[bh][j] = fmaxf(s, 0.0f) * drop_mask(42u, (unsigned)t, (unsigned)b, (unsigned)j);
    }
    __syncthreads();
    for (int j = lt; j < PRE; j += 64) {
      float s = 0.0f;
      const float* wr = P.w2 + (size_t)j * PRE;
      for (int k = 0; k < PRE; ++k) s += s_p1[bh][k] * wr[k];
      s = fmaxf(s, 0.0f) * drop_mask(43u, (unsigned)t, (unsigned)b, (unsigned)j);
      P.A1[(size_t)b * K1 + j] = (_Float16)s;
    }
    // pq = att_h @ Wq^T
    for (int a0 = lt; a0 < ATTD; a0 += 64) {
      float s = 0.0f;
      const float* hr = P.atth + (size_t)b * DRNN;
      const float* wr = P.wq + (size_t)a0 * DRNN;
      for (int k = 0; k < DRNN; ++k) s += hr[k] * wr[k];
      s_pq[bh][a0] = s;
    }
    __syncthreads();
    // energies: e[t] = v . tanh(pq + ploc + pm) + vb
    for (int te = lt; te < TENC; te += 64) {
      float loc[LF];
#pragma unroll
      for (int f = 0; f < LF; ++f) {
        float s = 0.0f;
        for (int k = 0; k < LK; ++k) {
          int idx = te - (LK - 1) / 2 + k;
          float cw = (idx >= 0 && idx < TENC) ? s_cum[bh][idx] : 0.0f;
          s += s_conv[f * LK + k] * cw;
        }
        loc[f] = s;
      }
      float e = vb0;
      const float* pmr = P.pm + ((size_t)b * TENC + te) * ATTD;
      for (int a0 = 0; a0 < ATTD; ++a0) {
        float s = s_pq[bh][a0] + pmr[a0];
        const float* wl = &s_wloc[a0 * LF];
#pragma unroll
        for (int f = 0; f < LF; ++f) s += wl[f] * loc[f];
        e += s_v[a0] * tanhf(s);
      }
      if (P.mask[(size_t)b * TENC + te]) e = -__builtin_inff();
      s_e[bh][te] = e;
    }
    __syncthreads();
    // softmax over T_enc
    {
      float m = -3.4e38f;
      for (int te = lt; te < TENC; te += 64) m = fmaxf(m, s_e[bh][te]);
      s_red[bh][lt] = m;
      __syncthreads();
      if (lt == 0) {
        float mm = s_red[bh][0];
        for (int i = 1; i < 64; ++i) mm = fmaxf(mm, s_red[bh][i]);
        s_red[bh][0] = mm;
      }
      __syncthreads();
      float rowmax = s_red[bh][0];
      __syncthreads();
      float ssum = 0.0f;
      for (int te = lt; te < TENC; te += 64) {
        float ex = __expf(s_e[bh][te] - rowmax);
        s_e[bh][te] = ex; ssum += ex;
      }
      s_red[bh][lt] = ssum;
      __syncthreads();
      if (lt == 0) {
        float s2 = 0.0f;
        for (int i = 0; i < 64; ++i) s2 += s_red[bh][i];
        s_red[bh][0] = s2;
      }
      __syncthreads();
      float inv = 1.0f / s_red[bh][0];
      for (int te = lt; te < TENC; te += 64) {
        float w = s_e[bh][te] * inv;
        s_e[bh][te] = w;
        s_cum[bh][te] += w;
        P.out_align[((size_t)b * TDEC + t) * TENC + te] = w;
      }
    }
    __syncthreads();
    // ctx = w @ memory[b]   (t-outer loop => coalesced 2KB row reads)
    {
      float acc[8];
#pragma unroll
      for (int i = 0; i < 8; ++i) acc[i] = 0.0f;
      const float* mrow = P.memory + (size_t)b * TENC * ENCD + lt * 8;
      for (int te = 0; te < TENC; ++te) {
        float w = s_e[bh][te];
        const float* mp = mrow + (size_t)te * ENCD;
#pragma unroll
        for (int i = 0; i < 8; ++i) acc[i] += w * mp[i];
      }
#pragma unroll
      for (int i = 0; i < 8; ++i) {
        int d = lt * 8 + i;
        _Float16 c16 = (_Float16)acc[i];
        P.A1[(size_t)b * K1 + 256 + d]    = c16;
        P.Api[(size_t)b * KPI + 1024 + d] = c16;
      }
    }
    gbar(P.bar);

    // ================= Phase B: LSTM1 (A1=[p|ctx|att_h], K=1792) ============
    lstm_gates(P.A1, K1, K1, P.w1cat, P.b1ih, P.b1hh, c1,
               P.A1, K1, 768, P.A2, K2, 0, P.atth, wid, lane);
    gbar(P.bar);

    // ================= Phase C: LSTM2 (A2=[att_h|dec_h], K=2048) ============
    lstm_gates(P.A2, K2, K2, P.w2cat, P.b2ih, P.b2hh, c2,
               P.A2, K2, 1024, P.Api, KPI, 0, nullptr, wid, lane);
    gbar(P.bar);

    // ================= Phase D: projection + stop gate ======================
    if (wid < 10) {                    // 2 M-tiles x 5 N-tiles of 16x16
      int mt2 = wid / 5, nt = wid % 5;
      int ncol = nt * 16 + (lane & 15);
      v8f acc;
      float pb = P.projb[ncol];
#pragma unroll
      for (int r = 0; r < 8; ++r) acc[r] = pb;
      const _Float16* Arow = P.Api + (size_t)mt2 * 16 * KPI;
      for (int k0 = 0; k0 < KPI; k0 += 32) {
        v16h a   = load_a16(Arow, KPI, 0, k0, lane);
        v16h bfr = load_b16(P.projw, KPI, nt * 16, k0, lane);
        acc = __builtin_amdgcn_wmma_f32_16x16x32_f16(false, a, false, bfr, (short)0, acc, false, false);
      }
#pragma unroll
      for (int r = 0; r < 8; ++r) {
        int m = mt2 * 16 + ((lane >> 4) << 3) + r;
        P.out_mel[((size_t)m * TDEC + t) * NMELS + ncol] = acc[r];
      }
    }
    if (blockIdx.x == NWG - 1) {       // stop gate: 1 output per batch row
      int m = tid >> 2, q = tid & 3;
      float s = 0.0f;
      const _Float16* pr = P.Api + (size_t)m * KPI + q * 384;
      const _Float16* gr = P.gatew + q * 384;
      for (int k = 0; k < 384; ++k) s += (float)pr[k] * (float)gr[k];
      s_gred[tid] = s;
      __syncthreads();
      if (q == 0)
        P.out_gate[(size_t)m * TDEC + t] =
            s_gred[tid] + s_gred[tid + 1] + s_gred[tid + 2] + s_gred[tid + 3] + P.gateb[0];
    }
    gbar(P.bar);
  }
}

// ---- setup kernels ---------------------------------------------------------
__global__ void k_zero_u32(unsigned* p, int n) {
  int i = blockIdx.x * blockDim.x + threadIdx.x;
  if (i < n) p[i] = 0u;
}
__global__ void k_f32_to_f16(const float* s, _Float16* d, size_t n) {
  for (size_t i = blockIdx.x * (size_t)blockDim.x + threadIdx.x; i < n;
       i += (size_t)gridDim.x * blockDim.x)
    d[i] = (_Float16)s[i];
}
__global__ void k_build_cat(const float* wih, const float* whh, _Float16* d,
                            int kih, int khh, int nrows) {
  int K = kih + khh;
  size_t total = (size_t)nrows * K;
  for (size_t i = blockIdx.x * (size_t)blockDim.x + threadIdx.x; i < total;
       i += (size_t)gridDim.x * blockDim.x) {
    int n = (int)(i / K), k = (int)(i % K);
    float v = (k < kih) ? wih[(size_t)n * kih + k] : whh[(size_t)n * khh + (k - kih)];
    d[i] = (_Float16)v;
  }
}

extern "C" void kernel_launch(void* const* d_in, const int* in_sizes, int n_in,
                              void* d_out, int out_size, void* d_ws, size_t ws_size,
                              hipStream_t stream) {
  (void)in_sizes; (void)n_in; (void)out_size; (void)ws_size;
  const float* memory = (const float*)d_in[0];
  const float* dec_in = (const float*)d_in[1];
  const unsigned char* mask = (const unsigned char*)d_in[2];
  const float* w1    = (const float*)d_in[3];
  const float* w2    = (const float*)d_in[4];
  const float* wq    = (const float*)d_in[5];
  const float* wm    = (const float*)d_in[6];
  const float* convw = (const float*)d_in[7];
  const float* wloc  = (const float*)d_in[8];
  const float* vw    = (const float*)d_in[9];
  const float* vb    = (const float*)d_in[10];
  const float* l1wih = (const float*)d_in[11];
  const float* l1whh = (const float*)d_in[12];
  const float* b1ih  = (const float*)d_in[13];
  const float* b1hh  = (const float*)d_in[14];
  const float* l2wih = (const float*)d_in[15];
  const float* l2whh = (const float*)d_in[16];
  const float* b2ih  = (const float*)d_in[17];
  const float* b2hh  = (const float*)d_in[18];
  const float* projw = (const float*)d_in[19];
  const float* projb = (const float*)d_in[20];
  const float* gatew = (const float*)d_in[21];
  const float* gateb = (const float*)d_in[22];

  char* ws = (char*)d_ws;
  size_t off = 0;
  auto alloc = [&](size_t bytes) {
    void* p = ws + off;
    off += (bytes + 255) & ~(size_t)255;
    return p;
  };
  _Float16* memf16  = (_Float16*)alloc((size_t)B_ * TENC * ENCD * 2);
  _Float16* w1cat   = (_Float16*)alloc((size_t)4096 * K1 * 2);
  _Float16* w2cat   = (_Float16*)alloc((size_t)4096 * K2 * 2);
  _Float16* projw16 = (_Float16*)alloc((size_t)80 * KPI * 2);
  _Float16* gatew16 = (_Float16*)alloc((size_t)KPI * 2);
  _Float16* attwm16 = (_Float16*)alloc((size_t)ATTD * ENCD * 2);
  float*    pm      = (float*)alloc((size_t)B_ * TENC * ATTD * 4);
  _Float16* A1      = (_Float16*)alloc((size_t)B_ * K1 * 2);
  _Float16* A2      = (_Float16*)alloc((size_t)B_ * K2 * 2);
  _Float16* Api     = (_Float16*)alloc((size_t)B_ * KPI * 2);
  float*    atth    = (float*)alloc((size_t)B_ * DRNN * 4);
  unsigned* bar     = (unsigned*)alloc(256);

  k_zero_u32<<<1, 64, 0, stream>>>(bar, 64);
  k_f32_to_f16<<<2048, 256, 0, stream>>>(memory, memf16, (size_t)B_ * TENC * ENCD);
  k_f32_to_f16<<<64, 256, 0, stream>>>(wm, attwm16, (size_t)ATTD * ENCD);
  k_f32_to_f16<<<64, 256, 0, stream>>>(projw, projw16, (size_t)80 * KPI);
  k_f32_to_f16<<<8, 256, 0, stream>>>(gatew, gatew16, (size_t)KPI);
  k_build_cat<<<2048, 256, 0, stream>>>(l1wih, l1whh, w1cat, 768, 1024, 4096);
  k_build_cat<<<2048, 256, 0, stream>>>(l2wih, l2whh, w2cat, 1024, 1024, 4096);

  DecParams P;
  P.memory = memory; P.dec_in = dec_in; P.mask = mask;
  P.w1 = w1; P.w2 = w2; P.wq = wq; P.convw = convw; P.wloc = wloc; P.vw = vw; P.vb = vb;
  P.b1ih = b1ih; P.b1hh = b1hh; P.b2ih = b2ih; P.b2hh = b2hh; P.projb = projb; P.gateb = gateb;
  P.w1cat = w1cat; P.w2cat = w2cat; P.projw = projw16; P.gatew = gatew16;
  P.memf16 = memf16; P.attwm = attwm16;
  P.pm = pm; P.atth = atth; P.A1 = A1; P.A2 = A2; P.Api = Api; P.bar = bar;
  float* out = (float*)d_out;
  P.out_mel   = out;
  P.out_gate  = out + (size_t)B_ * TDEC * NMELS;
  P.out_align = out + (size_t)B_ * TDEC * NMELS + (size_t)B_ * TDEC;

  decoder_persistent<<<NWG, BLK, 0, stream>>>(P);
}
